// gnn_decoder_82592221102353
// MI455X (gfx1250) — matrix operand order repaired and checked by reading, not compiled, not run on hardware
//
#include <hip/hip_runtime.h>
#include <math.h>

// Problem constants (must match reference)
#define BB 4
#define NN 1024
#define DD 64
#define TT 12
#define KO 5
#define ITERS 10

typedef __attribute__((ext_vector_type(16))) _Float16 v16h;
typedef __attribute__((ext_vector_type(8)))  _Float16 v8h;
typedef __attribute__((ext_vector_type(8)))  float    v8f;
typedef __attribute__((ext_vector_type(4)))  float    v4f;

union V16 { v16h v; struct { v8h lo; v8h hi; } p; };

// ---- WMMA fragment loaders (contiguous k = 16*(lane/16) + s, wave32) ----
__device__ __forceinline__ v16h frag_a(const _Float16* base, int ld) {
  int lane = threadIdx.x & 31;
  int row  = lane & 15;
  int koff = (lane >> 4) << 4;        // 0 or 16
  const _Float16* r = base + row * ld;
  V16 u;
  u.p.lo = *(const v8h*)(r + koff);
  u.p.hi = *(const v8h*)(r + 8 + koff);
  return u.v;
}
// B fragment 32x16 f16 from TRANSPOSED storage Bt[n][k], leading dim ld
__device__ __forceinline__ v16h frag_bt(const _Float16* base, int ld) {
  int lane = threadIdx.x & 31;
  int n  = lane & 15;
  int kb = (lane >> 4) << 4;          // 0 or 16
  const _Float16* r = base + n * ld + kb;
  V16 u;
  u.p.lo = *(const v8h*)(r);
  u.p.hi = *(const v8h*)(r + 8);
  return u.v;
}

__device__ __forceinline__ v8f wmma_f16(v16h a, v16h b, v8f c) {
  return __builtin_amdgcn_wmma_f32_16x16x32_f16(
      false, a, false, b, (short)0, c, false, false);
}

__device__ __forceinline__ v8f mm64(const _Float16* A, const _Float16* Bt,
                                    int tr, int tc, v8f acc) {
#pragma unroll
  for (int ks = 0; ks < 64; ks += 32) {
    v16h a  = frag_a (A  + (tr * 16) * 64 + ks, 64);
    v16h bt = frag_bt(Bt + (tc * 16) * 64 + ks, 64);
    acc = wmma_f16(a, bt, acc);
  }
  return acc;
}

__device__ __forceinline__ float sigmoidf_(float x) {
  return 1.0f / (1.0f + __expf(-x));
}

// ---- CDNA5 async global->LDS (per-lane b128), tracked by ASYNCcnt ----
// Generic pointer to an LDS object carries the LDS byte offset in its low
// 32 bits (flat aperture mapping).
__device__ __forceinline__ void async_ld_b128(const _Float16* gsrc,
                                              const _Float16* ldst) {
  unsigned loff = (unsigned)(size_t)ldst;
  asm volatile("global_load_async_to_lds_b128 %0, %1, off"
               :: "v"(loff), "v"(gsrc) : "memory");
}
__device__ __forceinline__ void wait_async_le3() {
#if __has_builtin(__builtin_amdgcn_s_wait_asynccnt)
  __builtin_amdgcn_s_wait_asynccnt(3);
#else
  asm volatile("s_wait_asynccnt 0x3" ::: "memory");
#endif
}
__device__ __forceinline__ void wait_async_le0() {
#if __has_builtin(__builtin_amdgcn_s_wait_asynccnt)
  __builtin_amdgcn_s_wait_asynccnt(0);
#else
  asm volatile("s_wait_asynccnt 0x0" ::: "memory");
#endif
}

// ---------------- one-time prep kernels ----------------

__global__ void __launch_bounds__(256) prep_edges(const float* __restrict__ src,
                                                  _Float16* __restrict__ dst) {
  size_t gid = (size_t)blockIdx.x * 256 + threadIdx.x;
  size_t i8 = gid * 8;
  v4f f0 = *(const v4f*)(src + i8);
  v4f f1 = *(const v4f*)(src + i8 + 4);
  v8h o;
  o[0] = (_Float16)f0[0]; o[1] = (_Float16)f0[1];
  o[2] = (_Float16)f0[2]; o[3] = (_Float16)f0[3];
  o[4] = (_Float16)f1[0]; o[5] = (_Float16)f1[1];
  o[6] = (_Float16)f1[2]; o[7] = (_Float16)f1[3];
  *(v8h*)(dst + i8) = o;
}

__global__ void __launch_bounds__(256) prep_h(const float* __restrict__ src,
                                              float* __restrict__ dst) {
  size_t gid = (size_t)blockIdx.x * 256 + threadIdx.x;
  *(v4f*)(dst + gid * 4) = *(const v4f*)(src + gid * 4);
}

__global__ void __launch_bounds__(256) prep_weights(
    const float* __restrict__ Wa,
    const float* __restrict__ Wz, const float* __restrict__ Wr,
    const float* __restrict__ Wh, const float* __restrict__ Uz,
    const float* __restrict__ Ur, const float* __restrict__ Uh,
    _Float16* __restrict__ WaT, _Float16* __restrict__ W6T) {
  int gid = blockIdx.x * 256 + threadIdx.x;     // grid covers 73728 exactly
  const int totalA = TT * DD * DD;              // 49152
  if (gid < totalA) {
    int t = gid >> 12, r = gid & 4095, e = r >> 6, d = r & 63;
    WaT[gid] = (_Float16)Wa[(t << 12) + (d << 6) + e];
  } else {
    int i = gid - totalA;                        // < 24576
    const float* Ws[6] = {Wz, Wr, Wh, Uz, Ur, Uh};
    int wi = i >> 12, r = i & 4095, e = r >> 6, d = r & 63;
    W6T[i] = (_Float16)Ws[wi][(d << 6) + e];
  }
}

// ---------------- per-iteration kernels ----------------

// hw_t[b][t][e][m] = (h[b] @ Wa[t])^T in f16. Block: 32 rows of m, all t.
__global__ void __launch_bounds__(256) hw_kernel(
    const float* __restrict__ h, const _Float16* __restrict__ WaT,
    _Float16* __restrict__ hwT) {
  __shared__ __align__(16) _Float16 sh_h[32 * 64];   // A tile [m][d]
  __shared__ __align__(16) _Float16 sh_w[64 * 64];   // Bt tile [e][d]
  __shared__ __align__(16) _Float16 sh_o[64 * 32];   // out staging [e][m]
  int b  = blockIdx.x >> 5;
  int m0 = (blockIdx.x & 31) * 32;
  int tid = threadIdx.x;
  int lane = tid & 31;
  int w = tid >> 5, tr = w >> 2, tc = w & 3;
  int nl = lane & 15, mb = (lane >> 4) << 3;

  const float* hsrc = h + ((size_t)b * NN + m0) * DD;
#pragma unroll
  for (int j = 0; j < 2; ++j) {
    int i = tid + j * 256;                        // 512 quads
    v4f f = *(const v4f*)(hsrc + i * 4);
    _Float16* d = &sh_h[i * 4];
    d[0] = (_Float16)f[0]; d[1] = (_Float16)f[1];
    d[2] = (_Float16)f[2]; d[3] = (_Float16)f[3];
  }
#pragma unroll 1
  for (int t = 0; t < TT; ++t) {
    __syncthreads();
    const v8h* wsrc = (const v8h*)(WaT + t * DD * DD);
    v8h* wdst = (v8h*)sh_w;
    wdst[tid] = wsrc[tid];
    wdst[tid + 256] = wsrc[tid + 256];
    __syncthreads();
    v8f acc = {};
#pragma unroll
    for (int ks = 0; ks < 64; ks += 32) {
      v16h a  = frag_a (&sh_h[(tr * 16) * 64 + ks], 64);
      v16h bt = frag_bt(&sh_w[(tc * 16) * 64 + ks], 64);
      acc = wmma_f16(a, bt, acc);
    }
    int e = tc * 16 + nl;
#pragma unroll
    for (int v = 0; v < 8; ++v) {
      int m = tr * 16 + mb + v;
      sh_o[e * 32 + m] = (_Float16)acc[v];
    }
    __syncthreads();
    int re = tid >> 2, seg = tid & 3;             // 64 rows x 4 segs
    v8h val = *(const v8h*)&sh_o[re * 32 + seg * 8];
    *(v8h*)(hwT + (((size_t)b * TT + t) * DD + re) * NN + m0 + seg * 8) = val;
  }
}

// Fused aggregation + GRU with triple-buffered async global->LDS pipeline.
__global__ void __launch_bounds__(256) agg_gru_kernel(
    const _Float16* __restrict__ e16, const _Float16* __restrict__ hwT,
    const float* __restrict__ ba, const _Float16* __restrict__ W6T,
    float* __restrict__ h) {
  __shared__ __align__(16) _Float16 sAbuf[3][32 * 64]; // edges tiles [n][m]
  __shared__ __align__(16) _Float16 sBbuf[3][64 * 64]; // hw_t tiles [e][m]
  __shared__ __align__(16) _Float16 sw[6 * 64 * 64];   // GRU weights (T)
  __shared__ __align__(16) float    sa32[32 * 64];
  __shared__ __align__(16) _Float16 sa16[32 * 64];
  __shared__ __align__(16) float    sh32[32 * 64];
  __shared__ __align__(16) _Float16 sh16[32 * 64];
  __shared__ __align__(16) _Float16 srh[32 * 64];

  int b  = blockIdx.x >> 5;
  int n0 = (blockIdx.x & 31) * 32;
  int tid = threadIdx.x;
  int lane = tid & 31;
  int w = tid >> 5, tr = w >> 2, tc = w & 3;
  int nl = lane & 15, mb = (lane >> 4) << 3;

  // stage six GRU weight matrices via async loads (retired by round-0 wait)
  {
#pragma unroll
    for (int j = 0; j < 12; ++j) {
      int i = tid + j * 256;
      async_ld_b128(W6T + i * 8, &sw[i * 8]);
    }
  }

  const int R = TT * 16;                       // 192 staging rounds
  // per-round staging: 1 b128 (edges) + 2 b128 (hw_t) per thread, all async
  auto stage_round = [&](int r, _Float16* bufA, _Float16* bufB) {
    int t = r >> 4, mt = r & 15, m0 = mt * 64;
    const _Float16* eb = e16 + (((size_t)b * TT + t) * NN + n0) * NN;
    const _Float16* hb = hwT + ((size_t)b * TT + t) * DD * NN;
    int rr = tid >> 3, seg = tid & 7;
    async_ld_b128(eb + (size_t)rr * NN + m0 + seg * 8, &bufA[rr * 64 + seg * 8]);
    async_ld_b128(hb + (size_t)rr * NN + m0 + seg * 8, &bufB[rr * 64 + seg * 8]);
    int i2 = tid + 256, r3 = i2 >> 3, s3 = i2 & 7;
    async_ld_b128(hb + (size_t)r3 * NN + m0 + s3 * 8, &bufB[r3 * 64 + s3 * 8]);
  };

  v8f acc = {};
  auto round_fn = [&](int r, _Float16* curA, _Float16* curB,
                      _Float16* nxtA, _Float16* nxtB) {
    if (r < R - 1) wait_async_le3();   // round r done; r+1 may be in flight
    else           wait_async_le0();
    __syncthreads();                   // r's data visible; slot (r+2)%3 free
    if (r + 2 < R) stage_round(r + 2, nxtA, nxtB);
#pragma unroll
    for (int ks = 0; ks < 64; ks += 32) {
      v16h a  = frag_a (&curA[(tr * 16) * 64 + ks], 64);
      v16h bt = frag_bt(&curB[(tc * 16) * 64 + ks], 64);
      acc = wmma_f16(a, bt, acc);
    }
  };

  // prologue: issue rounds 0 and 1
  stage_round(0, sAbuf[0], sBbuf[0]);
  stage_round(1, sAbuf[1], sBbuf[1]);
#pragma unroll 1
  for (int rr = 0; rr < R; rr += 3) {
    round_fn(rr + 0, sAbuf[0], sBbuf[0], sAbuf[2], sBbuf[2]);
    round_fn(rr + 1, sAbuf[1], sBbuf[1], sAbuf[0], sBbuf[0]);
    round_fn(rr + 2, sAbuf[2], sBbuf[2], sAbuf[1], sBbuf[1]);
  }
  __syncthreads();

  // ---- a = acc + ba; stage a (f32+f16) and h (f32+f16) in LDS ----
  {
    int e = tc * 16 + nl;
    float bav = ba[e];
#pragma unroll
    for (int v = 0; v < 8; ++v) {
      int m = tr * 16 + mb + v;
      float x = acc[v] + bav;
      sa32[m * 64 + e] = x;
      sa16[m * 64 + e] = (_Float16)x;
    }
  }
  {
    const float* hsrc = h + ((size_t)b * NN + n0) * DD;
#pragma unroll
    for (int j = 0; j < 2; ++j) {
      int i = tid + j * 256;
      v4f f = *(const v4f*)(hsrc + i * 4);
      *(v4f*)&sh32[i * 4] = f;
      _Float16* d = &sh16[i * 4];
      d[0] = (_Float16)f[0]; d[1] = (_Float16)f[1];
      d[2] = (_Float16)f[2]; d[3] = (_Float16)f[3];
    }
  }
  __syncthreads();

  // ---- GRU gates via WMMA: z, r ----
  v8f zacc = {};
  zacc = mm64(sa16, &sw[0 * 4096], tr, tc, zacc);   // a @ Wz
  zacc = mm64(sh16, &sw[3 * 4096], tr, tc, zacc);   // h @ Uz
  v8f racc = {};
  racc = mm64(sa16, &sw[1 * 4096], tr, tc, racc);   // a @ Wr
  racc = mm64(sh16, &sw[4 * 4096], tr, tc, racc);   // h @ Ur

  v8f z;
  {
    int e = tc * 16 + nl;
#pragma unroll
    for (int v = 0; v < 8; ++v) {
      int m = tr * 16 + mb + v;
      z[v] = sigmoidf_(zacc[v]);
      float rv = sigmoidf_(racc[v]);
      srh[m * 64 + e] = (_Float16)(rv * sh32[m * 64 + e]);
    }
  }
  __syncthreads();

  // ---- hh = tanh(a@Wh + (r*h)@Uh); h_new = (1-z)h + z*hh ----
  v8f hacc = {};
  hacc = mm64(sa16, &sw[2 * 4096], tr, tc, hacc);   // a @ Wh
  hacc = mm64(srh,  &sw[5 * 4096], tr, tc, hacc);   // rh @ Uh
  {
    int e = tc * 16 + nl;
    float* hdst = h + ((size_t)b * NN + n0) * DD;
#pragma unroll
    for (int v = 0; v < 8; ++v) {
      int m = tr * 16 + mb + v;
      float hh = tanhf(hacc[v]);
      float zz = z[v];
      float hold = sh32[m * 64 + e];
      hdst[(size_t)m * DD + e] = (1.0f - zz) * hold + zz * hh;
    }
  }
}

// ---------------- final logits + log_softmax ----------------
__global__ void __launch_bounds__(256) out_kernel(
    const float* __restrict__ h, const float* __restrict__ Wout,
    const float* __restrict__ bout, float* __restrict__ out) {
  int row = blockIdx.x * 256 + threadIdx.x;    // 0..4095 = (b,n)
  const float* hp = h + (size_t)row * DD;
  float l[KO];
#pragma unroll
  for (int k = 0; k < KO; ++k) l[k] = bout[k];
#pragma unroll 4
  for (int d = 0; d < DD; ++d) {
    float hv = hp[d];
#pragma unroll
    for (int k = 0; k < KO; ++k) l[k] += hv * Wout[d * KO + k];
  }
  float m = l[0];
#pragma unroll
  for (int k = 1; k < KO; ++k) m = fmaxf(m, l[k]);
  float s = 0.0f;
#pragma unroll
  for (int k = 0; k < KO; ++k) s += __expf(l[k] - m);
  float lse = __logf(s);
#pragma unroll
  for (int k = 0; k < KO; ++k) out[(size_t)row * KO + k] = l[k] - m - lse;
}

// ---------------- launcher ----------------
extern "C" void kernel_launch(void* const* d_in, const int* in_sizes, int n_in,
                              void* d_out, int out_size, void* d_ws,
                              size_t ws_size, hipStream_t stream) {
  (void)in_sizes; (void)n_in; (void)out_size; (void)ws_size;
  const float* x     = (const float*)d_in[0];
  const float* edges = (const float*)d_in[1];
  const float* Wa    = (const float*)d_in[2];
  const float* ba    = (const float*)d_in[3];
  const float* Wz    = (const float*)d_in[4];
  const float* Wr    = (const float*)d_in[5];
  const float* Wh    = (const float*)d_in[6];
  const float* Uz    = (const float*)d_in[7];
  const float* Ur    = (const float*)d_in[8];
  const float* Uh    = (const float*)d_in[9];
  const float* Wout  = (const float*)d_in[10];
  const float* bout  = (const float*)d_in[11];

  const size_t SZ_E16 = (size_t)BB * TT * NN * NN * 2;   // 100663296
  const size_t SZ_HWT = (size_t)BB * TT * DD * NN * 2;   //   6291456
  const size_t SZ_H   = (size_t)BB * NN * DD * 4;        //   1048576
  const size_t SZ_WAT = (size_t)TT * DD * DD * 2;        //     98304
  char* ws = (char*)d_ws;
  _Float16* e16 = (_Float16*)(ws);
  _Float16* hwT = (_Float16*)(ws + SZ_E16);
  float*    hbuf = (float*)(ws + SZ_E16 + SZ_HWT);
  _Float16* WaT = (_Float16*)(ws + SZ_E16 + SZ_HWT + SZ_H);
  _Float16* W6T = (_Float16*)(ws + SZ_E16 + SZ_HWT + SZ_H + SZ_WAT);

  prep_edges<<<24576, 256, 0, stream>>>(edges, e16);
  prep_h<<<256, 256, 0, stream>>>(x, hbuf);
  prep_weights<<<288, 256, 0, stream>>>(Wa, Wz, Wr, Wh, Uz, Ur, Uh, WaT, W6T);

  for (int it = 0; it < ITERS; ++it) {
    hw_kernel<<<BB * (NN / 32), 256, 0, stream>>>(hbuf, WaT, hwT);
    agg_gru_kernel<<<BB * (NN / 32), 256, 0, stream>>>(e16, hwT, ba, W6T, hbuf);
  }

  out_kernel<<<(BB * NN) / 256, 256, 0, stream>>>(hbuf, Wout, bout,
                                                  (float*)d_out);
}
